// Edge_aware_unit_70978629534273
// MI455X (gfx1250) — matrix operand order repaired and checked by reading, not compiled, not run on hardware
//
#include <hip/hip_runtime.h>

typedef __bf16 bf16;
typedef __attribute__((ext_vector_type(16))) __bf16 v16bf;
typedef __attribute__((ext_vector_type(8)))  __bf16 v8bf;
typedef __attribute__((ext_vector_type(8)))  float  v8f;
typedef __attribute__((ext_vector_type(4)))  float  v4f;

#define HW    25600
#define IMG_H 160
#define IMG_W 160
#define NIMG  8
#define NPIX  (NIMG * HW)          // 204800 pixels total
#define EPSBN 1e-5f

// ---------------- workspace layout (bytes) ----------------
// all activations NHWC bf16: [NIMG][160*160][C]
static const size_t OFF_XBF  = 0;            // 13,107,200 bf16 = 26,214,400 B
static const size_t OFF_C1   = 26214400ull;  // 26,214,400 bf16 = 52,428,800 B
static const size_t OFF_C2   = 78643200ull;  // 13,107,200 bf16 = 26,214,400 B
static const size_t OFF_EDGE = 104857600ull; // 13,107,200 bf16
static const size_t OFF_TMP  = 131072000ull; // 13,107,200 bf16 (c3raw then skipraw)
static const size_t OFF_WTS  = 157286400ull; // 155,648 bf16
static const size_t OFF_STAT = 157597696ull; // stats + scale/shift + zero pad buf

// ---------------- small utility kernels ----------------
__global__ void k_zero_f32(float* p, int n) {
    int i = blockIdx.x * blockDim.x + threadIdx.x;
    if (i < n) p[i] = 0.0f;
}

// NCHW f32 -> NHWC bf16; each thread emits 8 channels of one pixel (16B store)
__global__ void k_cvt_nhwc8(const float* __restrict__ x, bf16* __restrict__ y, int n8) {
    int i = blockIdx.x * blockDim.x + threadIdx.x;
    if (i < n8) {
        int chb = (i & 7) * 8;
        int pix = i >> 3;                 // n*HW + p
        int img = pix / HW;
        int p   = pix - img * HW;
        const float* src = x + ((size_t)img * 64 + chb) * HW + p;
        v8bf o;
#pragma unroll
        for (int e = 0; e < 8; ++e) o[e] = (bf16)src[(size_t)e * HW];
        ((v8bf*)y)[i] = o;
    }
}

// Re-order weights into B[co][rs*CIN + ci] (bf16). w1 is 1x1 so it is a flat copy.
__global__ void k_prep_w(const float* __restrict__ w1, const float* __restrict__ w2,
                         const float* __restrict__ w3, const float* __restrict__ w4,
                         bf16* __restrict__ wb) {
    int i = blockIdx.x * blockDim.x + threadIdx.x;
    if (i < 8192) {                                   // w1: [128][64] -> same order
        wb[i] = (bf16)w1[i];
    } else if (i < 8192 + 73728) {                    // w2: [64][128][3][3], K=1152
        int j = i - 8192;
        int co = j / 1152, rem = j % 1152;
        int rs = rem / 128, ci = rem % 128;
        wb[i] = (bf16)w2[(co * 128 + ci) * 9 + rs];
    } else if (i < 8192 + 73728 + 36864) {            // w3: [64][64][3][3], K=576
        int j = i - (8192 + 73728);
        int co = j / 576, rem = j % 576;
        int rs = rem / 64, ci = rem % 64;
        wb[i] = (bf16)w3[(co * 64 + ci) * 9 + rs];
    } else if (i < 8192 + 73728 + 36864 + 36864) {    // w4
        int j = i - (8192 + 73728 + 36864);
        int co = j / 576, rem = j % 576;
        int rs = rem / 64, ci = rem % 64;
        wb[i] = (bf16)w4[(co * 64 + ci) * 9 + rs];
    }
}

// ---------------- implicit-GEMM conv with bf16 WMMA (NHWC) ----------------
// M = MT x 16 pixels per wave, N = COUT (NTILES tiles), K = CIN*KS*KS.
// A = im2col activations (read once per pixel), B = weights (each B fragment
// feeds MT WMMAs -> weight bandwidth amortized by MT). Out-of-bounds taps read
// from a zeroed scratch buffer -> branch-free inner loop.
// in : [NIMG][HW][CIN] bf16   wt : [COUT][CIN*KS*KS] bf16 (k = rs*CIN + ci)
// out: [NIMG][HW][COUT] bf16 raw (pre-BN).  ssum/ssq: per-channel batch stats.
template <int CIN, int KS, int NTILES, int MT>
__global__ __launch_bounds__(256) void k_conv_wmma(
    const bf16* __restrict__ in, const bf16* __restrict__ wt,
    bf16* __restrict__ out, float* __restrict__ ssum, float* __restrict__ ssq,
    const bf16* __restrict__ zb) {
    constexpr int K    = CIN * KS * KS;
    constexpr int COUT = NTILES * 16;
    constexpr int PPW  = MT * 16;          // pixels per wave (HW % PPW == 0)

    const int tid  = threadIdx.x;
    const int lane = tid & 31;
    const int ln   = lane & 15;            // A row / B col index
    const int hh   = lane >> 4;            // half-wave
    const int wid  = blockIdx.x * 8 + (tid >> 5);
    const int pix0 = wid * PPW;            // wave's pixel base
    const int n    = pix0 / HW;
    const int ploc = pix0 - n * HW;
    const bf16* inN = in + (size_t)n * HW * CIN;

    int py[MT], px[MT];
#pragma unroll
    for (int mtt = 0; mtt < MT; ++mtt) {
        int p = ploc + mtt * 16 + ln;      // this lane's pixel in m-tile mtt
        py[mtt] = p / IMG_W;
        px[mtt] = p % IMG_W;
    }

    __shared__ float s_sum[COUT];
    __shared__ float s_sq[COUT];
    if (tid < COUT) { s_sum[tid] = 0.0f; s_sq[tid] = 0.0f; }
    __syncthreads();

    v8f acc[MT][NTILES];
#pragma unroll
    for (int mtt = 0; mtt < MT; ++mtt)
#pragma unroll
        for (int nt = 0; nt < NTILES; ++nt) acc[mtt][nt] = (v8f)0.0f;

#pragma unroll
    for (int rs = 0; rs < KS * KS; ++rs) {
        const bf16* pbA[MT];
#pragma unroll
        for (int mtt = 0; mtt < MT; ++mtt) {
            int iy = py[mtt], ix = px[mtt];
            bool ok = true;
            if (KS == 3) {
                int r = rs / 3, s = rs - r * 3;
                iy += r - 1; ix += s - 1;
                ok = ((unsigned)iy < (unsigned)IMG_H) &&
                     ((unsigned)ix < (unsigned)IMG_W);
            }
            // OOB lanes read zeros from the zero buffer (zero-padding)
            pbA[mtt] = ok ? (inN + (size_t)(iy * IMG_W + ix) * CIN) : zb;
        }
        const int kbase = rs * CIN;

#pragma unroll 1
        for (int cc = 0; cc < CIN; cc += 32) {
            // ---- A fragments: MT pixels x 32 channels (two 16B runs each) ----
            v16bf a[MT];
#pragma unroll
            for (int mtt = 0; mtt < MT; ++mtt) {
                v8bf alo = *(const v8bf*)(pbA[mtt] + cc + hh * 8);
                v8bf ahi = *(const v8bf*)(pbA[mtt] + cc + 16 + hh * 8);
                a[mtt] = __builtin_shufflevector(alo, ahi, 0, 1, 2, 3, 4, 5, 6, 7,
                                                 8, 9, 10, 11, 12, 13, 14, 15);
            }
#pragma unroll
            for (int nt = 0; nt < NTILES; ++nt) {
                // B fragment: weight row co = nt*16+ln, two 16B runs (k = hh*8.., 16+hh*8..)
                const bf16* pbB = wt + (size_t)(nt * 16 + ln) * K + kbase + cc + hh * 8;
                v8bf blo = *(const v8bf*)pbB;
                v8bf bhi = *(const v8bf*)(pbB + 16);
                v16bf b = __builtin_shufflevector(blo, bhi, 0, 1, 2, 3, 4, 5, 6, 7,
                                                  8, 9, 10, 11, 12, 13, 14, 15);
#pragma unroll
                for (int mtt = 0; mtt < MT; ++mtt) {    // one B feeds MT WMMAs
                    acc[mtt][nt] = __builtin_amdgcn_wmma_f32_16x16x32_bf16(
                        false, a[mtt], false, b, (short)0, acc[mtt][nt], false, false);
                }
            }
        }
    }

    // ---- epilogue: store raw NHWC + batch stats ----
    // lane holds D rows m=8*hh+rr (pixels), col n=nt*16+ln (channel)
#pragma unroll
    for (int nt = 0; nt < NTILES; ++nt) {
        const int co = nt * 16 + ln;
        float s = 0.0f, q = 0.0f;
#pragma unroll
        for (int mtt = 0; mtt < MT; ++mtt) {
#pragma unroll
            for (int rr = 0; rr < 8; ++rr) {
                float v = acc[mtt][nt][rr];
                s += v; q += v * v;
                int pp = ploc + mtt * 16 + hh * 8 + rr;
                out[((size_t)(n * HW + pp)) * COUT + co] = (bf16)v;
            }
        }
        s += __shfl_xor(s, 16, 32);        // partner half-wave: other 8 pixels
        q += __shfl_xor(q, 16, 32);
        if (hh == 0) {
            atomicAdd(&s_sum[co], s);      // LDS atomics: 8 waves/block
            atomicAdd(&s_sq[co], q);
        }
    }
    __syncthreads();
    if (tid < COUT) {                      // one global atomic per channel per block
        atomicAdd(&ssum[tid], s_sum[tid]);
        atomicAdd(&ssq[tid], s_sq[tid]);
    }
}

// ---------------- BN finalize: per-channel scale/shift ----------------
__global__ void k_finalize(const float* __restrict__ ssum, const float* __restrict__ ssq,
                           const float* __restrict__ g, const float* __restrict__ be,
                           float* __restrict__ scale, float* __restrict__ shift, int C) {
    int c = blockIdx.x * blockDim.x + threadIdx.x;
    if (c < C) {
        const float cnt = (float)NPIX;
        float m = ssum[c] / cnt;
        float v = ssq[c] / cnt - m * m;
        float sc = g[c] * rsqrtf(v + EPSBN);
        scale[c] = sc;
        shift[c] = be[c] - m * sc;
    }
}

// in-place BN+ReLU on NHWC bf16; 8 channels per thread (16B load/store)
__global__ void k_norm8(bf16* __restrict__ buf, const float* __restrict__ scale,
                        const float* __restrict__ shift, int cmask, int total8) {
    int i = blockIdx.x * blockDim.x + threadIdx.x;
    if (i < total8) {
        int chb = (i << 3) & cmask;
        v8bf v = ((const v8bf*)buf)[i];
        v4f s0 = *(const v4f*)(scale + chb);
        v4f s1 = *(const v4f*)(scale + chb + 4);
        v4f h0 = *(const v4f*)(shift + chb);
        v4f h1 = *(const v4f*)(shift + chb + 4);
#pragma unroll
        for (int e = 0; e < 4; ++e) {
            v[e]     = (bf16)fmaxf((float)v[e]     * s0[e] + h0[e], 0.0f);
            v[e + 4] = (bf16)fmaxf((float)v[e + 4] * s1[e] + h1[e], 0.0f);
        }
        ((v8bf*)buf)[i] = v;
    }
}

// BN+ReLU on NHWC raw [8][HW][64] -> f32 NCHW slice of concat out [8][128][HW]
// 4 pixels per thread: 16B coalesced f32 store, L2-cached bf16 gathers
__global__ void k_norm_out4(const bf16* __restrict__ raw, float* __restrict__ outp,
                            const float* __restrict__ scale, const float* __restrict__ shift,
                            int cbase, int total4) {
    int i = blockIdx.x * blockDim.x + threadIdx.x;
    if (i < total4) {
        int idx = i * 4;
        int p  = idx % HW;                 // HW%4==0 -> 4 pixels share (nn,c)
        int t  = idx / HW;
        int c  = t & 63;
        int nn = t >> 6;
        float sc = scale[c], sh = shift[c];
        const bf16* src = raw + ((size_t)(nn * HW + p)) * 64 + c;
        v4f o;
#pragma unroll
        for (int q = 0; q < 4; ++q)
            o[q] = fmaxf((float)src[(size_t)q * 64] * sc + sh, 0.0f);
        *(v4f*)(outp + ((size_t)nn * 128 + cbase + c) * HW + p) = o;
    }
}

// ------- fused BN+ReLU + combined depthwise 5x5 sobel, NHWC, 8 ch/thread -------
// Reads RAW c2, applies per-channel affine+ReLU to each tap on the fly
// (c2 is consumed only here -> saves a full normalize pass through L2).
__global__ void k_sobel8(const bf16* __restrict__ raw, bf16* __restrict__ dst,
                         const float* __restrict__ scale, const float* __restrict__ shift,
                         int total8) {
    int i = blockIdx.x * blockDim.x + threadIdx.x;
    if (i >= total8) return;
    int chb = (i & 7) * 8;
    int pix = i >> 3;                 // n*HW + p
    int n   = pix / HW;
    int p   = pix - n * HW;
    int y = p / IMG_W, x = p % IMG_W;
    const bf16* s = raw + ((size_t)n * HW) * 64 + chb;
    v4f s0 = *(const v4f*)(scale + chb);
    v4f s1 = *(const v4f*)(scale + chb + 4);
    v4f h0 = *(const v4f*)(shift + chb);
    v4f h1 = *(const v4f*)(shift + chb + 4);
    float acc[8];
#pragma unroll
    for (int e = 0; e < 8; ++e) acc[e] = 0.0f;
    auto T = [&](int dy, int dx, float w) {
        int yy = y + dy, xx = x + dx;
        if (((unsigned)yy < (unsigned)IMG_H) && ((unsigned)xx < (unsigned)IMG_W)) {
            v8bf v = *(const v8bf*)(s + (size_t)(yy * IMG_W + xx) * 64);
#pragma unroll
            for (int e = 0; e < 4; ++e) {
                acc[e]     += w * fmaxf((float)v[e]     * s0[e] + h0[e], 0.0f);
                acc[e + 4] += w * fmaxf((float)v[e + 4] * s1[e] + h1[e], 0.0f);
            }
        }
    };
    T(-1, -1, 4.0f); T(-1, 0, 4.0f);
    T(0, -2, -2.0f); T(0, -1, -8.0f); T(0, 1, 8.0f); T(0, 2, 2.0f);
    T(1, 0, -4.0f);  T(1, 1, -4.0f);
    v8bf o;
#pragma unroll
    for (int e = 0; e < 8; ++e) o[e] = (bf16)acc[e];
    ((v8bf*)dst)[i] = o;
}

// ---------------- host-side orchestration ----------------
extern "C" void kernel_launch(void* const* d_in, const int* in_sizes, int n_in,
                              void* d_out, int out_size, void* d_ws, size_t ws_size,
                              hipStream_t stream) {
    (void)in_sizes; (void)n_in; (void)out_size; (void)ws_size;
    const float* x   = (const float*)d_in[0];
    const float* w1  = (const float*)d_in[1];
    const float* g1  = (const float*)d_in[3];
    const float* be1 = (const float*)d_in[4];
    const float* w2  = (const float*)d_in[5];
    const float* g2  = (const float*)d_in[7];
    const float* be2 = (const float*)d_in[8];
    const float* w3  = (const float*)d_in[9];
    const float* g3  = (const float*)d_in[11];
    const float* be3 = (const float*)d_in[12];
    const float* w4  = (const float*)d_in[13];
    const float* g4  = (const float*)d_in[15];
    const float* be4 = (const float*)d_in[16];
    float* outp = (float*)d_out;

    char* ws = (char*)d_ws;
    bf16* xbf  = (bf16*)(ws + OFF_XBF);
    bf16* c1   = (bf16*)(ws + OFF_C1);
    bf16* c2   = (bf16*)(ws + OFF_C2);
    bf16* edge = (bf16*)(ws + OFF_EDGE);
    bf16* tmp  = (bf16*)(ws + OFF_TMP);
    bf16* wall = (bf16*)(ws + OFF_WTS);
    bf16* w1b = wall;
    bf16* w2b = wall + 8192;
    bf16* w3b = wall + 81920;
    bf16* w4b = wall + 118784;

    float* stat = (float*)(ws + OFF_STAT);
    float* sum0 = stat + 0 * 256, *sq0 = sum0 + 128;
    float* sum1 = stat + 1 * 256, *sq1 = sum1 + 128;
    float* sum2 = stat + 2 * 256, *sq2 = sum2 + 128;
    float* sum3 = stat + 3 * 256, *sq3 = sum3 + 128;
    float* sca  = stat + 1024;
    float* sc0 = sca + 0 * 256, *sh0 = sc0 + 128;
    float* sc1 = sca + 1 * 256, *sh1 = sc1 + 128;
    float* sc2 = sca + 2 * 256, *sh2 = sc2 + 128;
    float* sc3 = sca + 3 * 256, *sh3 = sc3 + 128;
    bf16*  zb  = (bf16*)(stat + 2048);     // 512B zeroed pad buffer (>= CIN elems)

    const int NX  = NIMG * 64 * HW;    // 13,107,200
    const int NC1 = NIMG * 128 * HW;   // 26,214,400

    // 0) zero stats + zero-pad buffer (scale region overwritten by finalize)
    k_zero_f32<<<9, 256, 0, stream>>>(stat, 2176);
    // 1) input NCHW f32 -> NHWC bf16; weights reordered to k = rs*CIN + ci
    k_cvt_nhwc8<<<(NX / 8) / 256, 256, 0, stream>>>(x, xbf, NX / 8);
    k_prep_w<<<608, 256, 0, stream>>>(w1, w2, w3, w4, wall);

    // 2) conv1 1x1 64->128 (WMMA): MT=2 -> 32 px/wave, 800 blocks
    k_conv_wmma<64, 1, 8, 2><<<800, 256, 0, stream>>>(xbf, w1b, c1, sum0, sq0, zb);
    k_finalize<<<1, 128, 0, stream>>>(sum0, sq0, g1, be1, sc0, sh0, 128);
    k_norm8<<<(NC1 / 8) / 256, 256, 0, stream>>>(c1, sc0, sh0, 127, NC1 / 8);

    // 3) conv2 3x3 128->64 (WMMA): MT=4 -> 64 px/wave, 400 blocks
    k_conv_wmma<128, 3, 4, 4><<<400, 256, 0, stream>>>(c1, w2b, c2, sum1, sq1, zb);
    k_finalize<<<1, 128, 0, stream>>>(sum1, sq1, g2, be2, sc1, sh1, 64);

    // 4) fused BN+ReLU + depthwise sobel (c2 raw -> edge)
    k_sobel8<<<(NX / 8) / 256, 256, 0, stream>>>(c2, edge, sc1, sh1, NX / 8);

    // 5) conv3 3x3 64->64 (WMMA) -> f32 output channels [64..128)
    k_conv_wmma<64, 3, 4, 4><<<400, 256, 0, stream>>>(edge, w3b, tmp, sum2, sq2, zb);
    k_finalize<<<1, 128, 0, stream>>>(sum2, sq2, g3, be3, sc2, sh2, 64);
    k_norm_out4<<<(NX / 4) / 256, 256, 0, stream>>>(tmp, outp, sc2, sh2, 64, NX / 4);

    // 6) skip conv 3x3 64->64 (WMMA) on input -> f32 output channels [0..64)
    k_conv_wmma<64, 3, 4, 4><<<400, 256, 0, stream>>>(xbf, w4b, tmp, sum3, sq3, zb);
    k_finalize<<<1, 128, 0, stream>>>(sum3, sq3, g4, be4, sc3, sh3, 64);
    k_norm_out4<<<(NX / 4) / 256, 256, 0, stream>>>(tmp, outp, sc3, sh3, 0, NX / 4);
}
